// Transformer_274877907081
// MI455X (gfx1250) — matrix operand order repaired and checked by reading, not compiled, not run on hardware
//
#include <hip/hip_runtime.h>
#include <cstdint>

typedef __bf16 bf16;
typedef __attribute__((ext_vector_type(16))) __bf16 bf16x16;
typedef __attribute__((ext_vector_type(8)))  float  f32x8;
typedef __attribute__((ext_vector_type(4)))  unsigned int u32x4;
typedef __attribute__((ext_vector_type(4)))  int v4i;

#if __has_builtin(__builtin_amdgcn_global_load_async_to_lds_b128) && \
    __has_builtin(__builtin_amdgcn_s_wait_asynccnt)
#define HAVE_ASYNC_LDS 1
typedef __attribute__((address_space(1))) v4i as1_v4i;
typedef __attribute__((address_space(3))) v4i as3_v4i;
#endif

union Frag { bf16x16 v; unsigned int u[8]; };

__device__ __forceinline__ f32x8 wmma_bf16f32(bf16x16 a, bf16x16 b, f32x8 c) {
  // emits v_wmma_f32_16x16x32_bf16
  return __builtin_amdgcn_wmma_f32_16x16x32_bf16(false, a, false, b, (short)0, c,
                                                 false, false);
}

__device__ __forceinline__ f32x8 zero8() {
  f32x8 z;
#pragma unroll
  for (int i = 0; i < 8; ++i) z[i] = 0.f;
  return z;
}

__device__ __forceinline__ float gelu_exact(float x) {
  return 0.5f * x * (1.0f + erff(x * 0.70710678118654752f));
}

// ds_swizzle_b32 with immediate XOR pattern: offset = {and=0x1f, or=0, xor=mask}
template <int IMM>
__device__ __forceinline__ float swz(float x) {
  return __builtin_bit_cast(
      float, __builtin_amdgcn_ds_swizzle(__builtin_bit_cast(int, x), IMM));
}

// Batched butterfly max over 16-lane halves for 8 independent values:
// per step, 8 independent ds_swizzle ops (clause + single wait), then 8 maxes.
template <int IMM>
__device__ __forceinline__ void bfly_max8(float (&m)[8]) {
  float t[8];
#pragma unroll
  for (int i = 0; i < 8; ++i) t[i] = swz<IMM>(m[i]);
#pragma unroll
  for (int i = 0; i < 8; ++i) m[i] = fmaxf(m[i], t[i]);
}

// Wave-local LDS ordering: DS ops from one wave complete in order (ISA 7.3);
// we only need to stop the compiler from reordering around the staging buffer.
__device__ __forceinline__ void wave_lds_fence() { asm volatile("" ::: "memory"); }

#define SCALE_ATTN 0.05103103630798288f  // 384^-0.5 (reference scales by full dim)

// ---------------------------------------------------------------------------
// fp32 [depth][K][N] -> bf16 [depth][N][K]  (transpose + narrow, once per call)
// ---------------------------------------------------------------------------
__global__ __launch_bounds__(256) void convt_kernel(const float* __restrict__ src,
                                                    bf16* __restrict__ dst,
                                                    int Kd, int Nd, int total) {
  int idx = blockIdx.x * 256 + threadIdx.x;
  if (idx >= total) return;
  int per = Kd * Nd;
  int d   = idx / per;
  int rem = idx - d * per;
  int n   = rem / Kd;
  int kk  = rem - n * Kd;
  dst[idx] = (bf16)src[(long)d * per + (long)kk * Nd + n];
}

// ---------------------------------------------------------------------------
// LayerNorm: one wave (32 lanes) per row of 384, 8 rows per block, bf16 out.
// ---------------------------------------------------------------------------
__global__ __launch_bounds__(256) void ln_kernel(const float* __restrict__ x,
                                                 const float* __restrict__ w,
                                                 const float* __restrict__ b,
                                                 bf16* __restrict__ out) {
  const int lane = threadIdx.x & 31, wid = threadIdx.x >> 5;
  const long row = (long)blockIdx.x * 8 + wid;
  const float* xr = x + row * 384;
  float vals[12];
  float s = 0.f, s2 = 0.f;
#pragma unroll
  for (int e = 0; e < 12; ++e) {
    float v = xr[lane + e * 32];
    vals[e] = v;
    s += v;
    s2 += v * v;
  }
  s += swz<0x7C01>(s); s2 += swz<0x7C01>(s2);
  s += swz<0x7C02>(s); s2 += swz<0x7C02>(s2);
  s += swz<0x7C04>(s); s2 += swz<0x7C04>(s2);
  s += swz<0x7C08>(s); s2 += swz<0x7C08>(s2);
  s += swz<0x7C10>(s); s2 += swz<0x7C10>(s2);
  const float inv = 1.0f / 384.0f;
  float mean = s * inv;
  float var  = s2 * inv - mean * mean;
  float r    = rsqrtf(var + 1e-5f);
#pragma unroll
  for (int e = 0; e < 12; ++e) {
    int c = lane + e * 32;
    out[row * 384 + c] = (bf16)((vals[e] - mean) * r * w[c] + b[c]);
  }
}

// ---------------------------------------------------------------------------
// GEMM: C[M,N] = A[M,K] * W[K,N] + bias, A bf16 row-major, Wt bf16 [N][K].
// Block tile 128x128 (128 threads / 4 waves), BLK_K=32; each wave 64x64
// (4x4 WMMA tiles -> 16 WMMAs per k-step).  Double-buffered LDS staging via
// GLOBAL_LOAD_ASYNC_TO_LDS_B128 (ASYNCcnt) when available: one barrier per
// k-step, next tile's loads in flight during the WMMAs.
// EPI 0: QKV scatter (Q pre-scaled by 384^-0.5; Q/K as [b,h,n,d], V as
//        [b,h,d,n] transposed), bf16.  part/head/dd-base are tile-uniform.
// EPI 1: fp32 out += acc + bias (residual accumulate in place)
// EPI 2: bf16 out = gelu(acc + bias)
// ---------------------------------------------------------------------------
template <int EPI>
__global__ __launch_bounds__(128) void gemm_kernel(
    const bf16* __restrict__ A, const bf16* __restrict__ Wt,
    const float* __restrict__ bias, int Kd, int Nt,
    float* __restrict__ outF, bf16* __restrict__ outB,
    bf16* __restrict__ qb, bf16* __restrict__ kb, bf16* __restrict__ vtb) {
  __shared__ bf16 sA[2][128 * 32];
  __shared__ bf16 sB[2][128 * 32];
  const int tid  = threadIdx.x;
  const int lane = tid & 31;
  const int wid  = tid >> 5;
  const int lh   = lane >> 4, l15 = lane & 15;
  const int m0   = blockIdx.y * 128;
  const int n0   = blockIdx.x * 128;
  const int wm0  = (wid >> 1) * 64;
  const int wn0  = (wid & 1) * 64;

  f32x8 acc[4][4];
#pragma unroll
  for (int mi = 0; mi < 4; ++mi)
#pragma unroll
    for (int ni = 0; ni < 4; ++ni) acc[mi][ni] = zero8();

  const int arow = tid >> 2, acol8 = (tid & 3) * 8;

  auto stage = [&](int buf, int kt) {
#pragma unroll
    for (int j = 0; j < 4; ++j) {
      int row = arow + j * 32;
#ifdef HAVE_ASYNC_LDS
      __builtin_amdgcn_global_load_async_to_lds_b128(
          (as1_v4i*)(void*)&A[(long)(m0 + row) * Kd + kt + acol8],
          (as3_v4i*)(void*)&sA[buf][row * 32 + acol8], 0, 0);
      __builtin_amdgcn_global_load_async_to_lds_b128(
          (as1_v4i*)(void*)&Wt[(long)(n0 + row) * Kd + kt + acol8],
          (as3_v4i*)(void*)&sB[buf][row * 32 + acol8], 0, 0);
#else
      *(u32x4*)&sA[buf][row * 32 + acol8] =
          *(const u32x4*)&A[(long)(m0 + row) * Kd + kt + acol8];
      *(u32x4*)&sB[buf][row * 32 + acol8] =
          *(const u32x4*)&Wt[(long)(n0 + row) * Kd + kt + acol8];
#endif
    }
  };

  stage(0, 0);

  const unsigned* sAu = (const unsigned*)&sA[0][0];
  const unsigned* sBu = (const unsigned*)&sB[0][0];

  for (int kt = 0; kt < Kd; kt += 32) {
    const int cur  = (kt >> 5) & 1;
    const int curo = cur * (128 * 32 / 2);  // dword offset of current buffer
#ifdef HAVE_ASYNC_LDS
    __builtin_amdgcn_s_wait_asynccnt(0);  // own async loads drained
#endif
    __syncthreads();  // all waves drained -> tile visible; prev reads done
    if (kt + 32 < Kd) stage(cur ^ 1, kt + 32);

    Frag af[4];
#pragma unroll
    for (int mi = 0; mi < 4; ++mi) {
      int row = wm0 + mi * 16 + l15;
#pragma unroll
      for (int i = 0; i < 8; ++i) {
        int off = 2 * i + lh * 16;
        af[mi].u[i] = sAu[curo + ((row * 32 + off) >> 1)];
      }
    }
#pragma unroll
    for (int ni = 0; ni < 4; ++ni) {
      Frag bfr;
      int coln = wn0 + ni * 16 + l15;
#pragma unroll
      for (int i = 0; i < 8; ++i)
        bfr.u[i] = sBu[curo + ((coln * 32 + 2 * i + lh * 16) >> 1)];
#pragma unroll
      for (int mi = 0; mi < 4; ++mi)
        acc[mi][ni] = wmma_bf16f32(af[mi].v, bfr.v, acc[mi][ni]);
    }
  }

  // ---- epilogue ----
  const int bb  = m0 >> 11;          // block never crosses a batch boundary
  const int nnb = (m0 & 2047) + wm0; // row-in-batch base for this wave
#pragma unroll
  for (int mi = 0; mi < 4; ++mi)
#pragma unroll
    for (int ni = 0; ni < 4; ++ni) {
      const int colt = n0 + wn0 + ni * 16;  // tile-uniform column base
      const int col  = colt + l15;
      const float bv = bias[col];
      const int rb   = nnb + mi * 16 + lh * 8;  // row-in-batch base (+i)
      if constexpr (EPI == 0) {
        // 16-wide tile never crosses a 64/384 boundary -> uniform routing
        const int part = colt / 384;
        const int rem  = colt - part * 384;
        const int hh   = rem >> 6;
        const int dd   = (rem & 63) + l15;
        if (part == 0) {
          bf16* p = qb + (((long)(bb * 6 + hh) * 2048) + rb) * 64 + dd;
#pragma unroll
          for (int i = 0; i < 8; ++i)
            p[i * 64] = (bf16)((acc[mi][ni][i] + bv) * SCALE_ATTN);
        } else if (part == 1) {
          bf16* p = kb + (((long)(bb * 6 + hh) * 2048) + rb) * 64 + dd;
#pragma unroll
          for (int i = 0; i < 8; ++i) p[i * 64] = (bf16)(acc[mi][ni][i] + bv);
        } else {
          bf16* p = vtb + (((long)(bb * 6 + hh) * 64) + dd) * 2048 + rb;
#pragma unroll
          for (int i = 0; i < 8; ++i) p[i] = (bf16)(acc[mi][ni][i] + bv);
        }
      } else if constexpr (EPI == 1) {
        float* p = outF + (long)(m0 + wm0 + mi * 16 + lh * 8) * Nt + col;
#pragma unroll
        for (int i = 0; i < 8; ++i) {
          float v = acc[mi][ni][i] + bv + p[(long)i * Nt];
          p[(long)i * Nt] = v;
        }
      } else {
        bf16* p = outB + (long)(m0 + wm0 + mi * 16 + lh * 8) * Nt + col;
#pragma unroll
        for (int i = 0; i < 8; ++i)
          p[(long)i * Nt] = (bf16)gelu_exact(acc[mi][ni][i] + bv);
      }
    }
}

// ---------------------------------------------------------------------------
// Flash attention: one wave per 16-query tile of one (b,h); 32 keys / iter.
// Scores = Qs(16x64) * K^T (Q pre-scaled), online softmax; the row-sum of P
// is accumulated by an extra WMMA against an all-ones B tile (accL), so only
// the row-max needs lane reductions (batched ds_swizzle butterflies).
// P staged C->A layout through wave-private LDS (per-wave DS ops in-order).
// acc += P(16x32) * V (4 WMMA over d).  V held transposed [b,h,d,n].
// ---------------------------------------------------------------------------
__global__ __launch_bounds__(128) void attn_kernel(const bf16* __restrict__ Q,
                                                   const bf16* __restrict__ Kb,
                                                   const bf16* __restrict__ Vt,
                                                   bf16* __restrict__ out) {
  __shared__ bf16 sP[4 * 16 * 32];
  const int tid  = threadIdx.x;
  const int lane = tid & 31, wid = tid >> 5;
  const int lh = lane >> 4, l15 = lane & 15;
  const int tile = blockIdx.x * 4 + wid;
  const int qt = tile & 127;
  const int hh = (tile >> 7) % 6;
  const int bb = tile / (128 * 6);
  const int q0 = qt * 16;

  const bf16* qp = Q  + (long)((bb * 6 + hh) * 2048) * 64;
  const bf16* kp = Kb + (long)((bb * 6 + hh) * 2048) * 64;
  const bf16* vp = Vt + (long)((bb * 6 + hh) * 64) * 2048;

  Frag qa0, qa1;
#pragma unroll
  for (int i = 0; i < 8; ++i) {
    int off = 2 * i + lh * 16;
    qa0.u[i] = *(const unsigned*)&qp[(q0 + l15) * 64 + off];
    qa1.u[i] = *(const unsigned*)&qp[(q0 + l15) * 64 + 32 + off];
  }

  Frag ones;  // all-ones bf16 B tile for fused row-sum WMMA
#pragma unroll
  for (int i = 0; i < 8; ++i) ones.u[i] = 0x3F803F80u;

  f32x8 acc[4];
  acc[0] = zero8(); acc[1] = zero8(); acc[2] = zero8(); acc[3] = zero8();
  f32x8 accL = zero8();  // softmax denominator per row (every column equal)
  float mrun[8];
#pragma unroll
  for (int i = 0; i < 8; ++i) mrun[i] = -3.0e38f;

  bf16* myP = sP + wid * (16 * 32);
  const unsigned* myPu = (const unsigned*)myP;

  for (int kb0 = 0; kb0 < 2048; kb0 += 32) {
    f32x8 s0 = zero8(), s1 = zero8();
    Frag kf;
#pragma unroll
    for (int i = 0; i < 8; ++i)
      kf.u[i] = *(const unsigned*)&kp[(kb0 + l15) * 64 + 2 * i + lh * 16];
    s0 = wmma_bf16f32(qa0.v, kf.v, s0);
#pragma unroll
    for (int i = 0; i < 8; ++i)
      kf.u[i] = *(const unsigned*)&kp[(kb0 + l15) * 64 + 32 + 2 * i + lh * 16];
    s0 = wmma_bf16f32(qa1.v, kf.v, s0);
#pragma unroll
    for (int i = 0; i < 8; ++i)
      kf.u[i] = *(const unsigned*)&kp[(kb0 + 16 + l15) * 64 + 2 * i + lh * 16];
    s1 = wmma_bf16f32(qa0.v, kf.v, s1);
#pragma unroll
    for (int i = 0; i < 8; ++i)
      kf.u[i] = *(const unsigned*)&kp[(kb0 + 16 + l15) * 64 + 32 + 2 * i + lh * 16];
    s1 = wmma_bf16f32(qa1.v, kf.v, s1);

    // ---- row max: batched butterflies (4 steps x 8 independent swizzles) ---
    float mx[8];
#pragma unroll
    for (int i = 0; i < 8; ++i) mx[i] = fmaxf(s0[i], s1[i]);
    bfly_max8<0x7C01>(mx);
    bfly_max8<0x7C02>(mx);
    bfly_max8<0x7C04>(mx);
    bfly_max8<0x7C08>(mx);

    wave_lds_fence();  // prior P-frag reads ordered before this iter's writes
#pragma unroll
    for (int i = 0; i < 8; ++i) {
      float mnew  = fmaxf(mrun[i], mx[i]);
      float alpha = __expf(mrun[i] - mnew);
      mrun[i] = mnew;
      float p0 = __expf(s0[i] - mnew);
      float p1 = __expf(s1[i] - mnew);
      acc[0][i] *= alpha; acc[1][i] *= alpha;
      acc[2][i] *= alpha; acc[3][i] *= alpha;
      accL[i]   *= alpha;
      int prow = i + lh * 8;
      myP[prow * 32 + l15]      = (bf16)p0;
      myP[prow * 32 + 16 + l15] = (bf16)p1;
    }
    wave_lds_fence();  // P writes ordered before A-fragment gather

    Frag pf;
#pragma unroll
    for (int i = 0; i < 8; ++i) {
      int off = 2 * i + lh * 16;
      pf.u[i] = myPu[(l15 * 32 + off) >> 1];
    }
    accL = wmma_bf16f32(pf.v, ones.v, accL);  // denominator += rowsum(P)
#pragma unroll
    for (int t = 0; t < 4; ++t) {
      Frag vf;
#pragma unroll
      for (int i = 0; i < 8; ++i)
        vf.u[i] = *(const unsigned*)&vp[(t * 16 + l15) * 2048 + kb0 + 2 * i + lh * 16];
      acc[t] = wmma_bf16f32(pf.v, vf.v, acc[t]);
    }
  }

#pragma unroll
  for (int t = 0; t < 4; ++t)
#pragma unroll
    for (int i = 0; i < 8; ++i) {
      int rq = q0 + i + lh * 8;
      int dd = t * 16 + l15;
      float o = acc[t][i] / accL[i];
      out[((long)(bb * 2048 + rq)) * 384 + hh * 64 + dd] = (bf16)o;
    }
}

// ---------------------------------------------------------------------------
extern "C" void kernel_launch(void* const* d_in, const int* in_sizes, int n_in,
                              void* d_out, int out_size, void* d_ws,
                              size_t ws_size, hipStream_t stream) {
  (void)in_sizes; (void)n_in; (void)out_size; (void)ws_size;
  const float* x_in  = (const float*)d_in[0];
  const float* ln1w  = (const float*)d_in[1];
  const float* ln1b  = (const float*)d_in[2];
  const float* qkvw  = (const float*)d_in[3];
  const float* qkvb  = (const float*)d_in[4];
  const float* projw = (const float*)d_in[5];
  const float* projb = (const float*)d_in[6];
  const float* ln2w  = (const float*)d_in[7];
  const float* ln2b  = (const float*)d_in[8];
  const float* w1    = (const float*)d_in[9];
  const float* b1    = (const float*)d_in[10];
  const float* w2    = (const float*)d_in[11];
  const float* b2    = (const float*)d_in[12];
  float* xr = (float*)d_out;  // residual stream lives in d_out

  char* ws = (char*)d_ws;
  size_t off = 0;
  auto carve = [&](size_t bytes) -> char* {
    char* p = ws + off;
    off = (off + bytes + 255) & ~(size_t)255;
    return p;
  };
  bf16* wt_qkv  = (bf16*)carve((size_t)6 * 1152 * 384 * 2);
  bf16* wt_proj = (bf16*)carve((size_t)6 * 384 * 384 * 2);
  bf16* wt_w1   = (bf16*)carve((size_t)6 * 1536 * 384 * 2);
  bf16* wt_w2   = (bf16*)carve((size_t)6 * 384 * 1536 * 2);
  bf16* xnorm   = (bf16*)carve((size_t)16384 * 384 * 2);
  bf16* qbuf    = (bf16*)carve((size_t)8 * 6 * 2048 * 64 * 2);
  bf16* kbuf    = (bf16*)carve((size_t)8 * 6 * 2048 * 64 * 2);
  bf16* vtbuf   = (bf16*)carve((size_t)8 * 6 * 64 * 2048 * 2);
  bf16* attno   = (bf16*)carve((size_t)16384 * 384 * 2);
  bf16* hid     = (bf16*)carve((size_t)16384 * 1536 * 2);

  (void)hipMemcpyAsync(xr, x_in, (size_t)16384 * 384 * 4,
                       hipMemcpyDeviceToDevice, stream);

  auto cdiv = [](int a, int b) { return (a + b - 1) / b; };
  convt_kernel<<<cdiv(6 * 384 * 1152, 256), 256, 0, stream>>>(qkvw, wt_qkv, 384, 1152, 6 * 384 * 1152);
  convt_kernel<<<cdiv(6 * 384 * 384, 256), 256, 0, stream>>>(projw, wt_proj, 384, 384, 6 * 384 * 384);
  convt_kernel<<<cdiv(6 * 384 * 1536, 256), 256, 0, stream>>>(w1, wt_w1, 384, 1536, 6 * 384 * 1536);
  convt_kernel<<<cdiv(6 * 1536 * 384, 256), 256, 0, stream>>>(w2, wt_w2, 1536, 384, 6 * 1536 * 384);

  for (int l = 0; l < 6; ++l) {
    ln_kernel<<<2048, 256, 0, stream>>>(xr, ln1w + l * 384, ln1b + l * 384, xnorm);
    gemm_kernel<0><<<dim3(9, 128), 128, 0, stream>>>(
        xnorm, wt_qkv + (size_t)l * 1152 * 384, qkvb + l * 1152, 384, 1152,
        nullptr, nullptr, qbuf, kbuf, vtbuf);
    attn_kernel<<<1536, 128, 0, stream>>>(qbuf, kbuf, vtbuf, attno);
    gemm_kernel<1><<<dim3(3, 128), 128, 0, stream>>>(
        attno, wt_proj + (size_t)l * 384 * 384, projb + l * 384, 384, 384, xr,
        nullptr, nullptr, nullptr, nullptr);
    ln_kernel<<<2048, 256, 0, stream>>>(xr, ln2w + l * 384, ln2b + l * 384, xnorm);
    gemm_kernel<2><<<dim3(12, 128), 128, 0, stream>>>(
        xnorm, wt_w1 + (size_t)l * 1536 * 384, b1 + l * 1536, 384, 1536,
        nullptr, hid, nullptr, nullptr, nullptr);
    gemm_kernel<1><<<dim3(3, 128), 128, 0, stream>>>(
        hid, wt_w2 + (size_t)l * 384 * 1536, b2 + l * 384, 1536, 384, xr,
        nullptr, nullptr, nullptr, nullptr);
  }
}
